// GGANN_83597243449444
// MI455X (gfx1250) — matrix operand order repaired and checked by reading, not compiled, not run on hardware
//
#include <hip/hip_runtime.h>
#include <hip/hip_bf16.h>
#include <math.h>

#define N_NODES 20000
#define N_EDGES 320000
#define N_GRAPH 64
#define D_IN    256
#define D_H1    512
#define D_H2    256
#define N_HEADS 2
#define N_CLS   16
#define N_STEPS 4
#define N_ETYPE 4

typedef __attribute__((ext_vector_type(16))) _Float16 v16h;
typedef __attribute__((ext_vector_type(8)))  _Float16 v8h;
typedef __attribute__((ext_vector_type(8)))  float    v8f;

#define HD static __device__ __forceinline__

HD float sigmoid_(float x) { return 1.0f / (1.0f + expf(-x)); }

// Low 32 bits of a generic pointer to __shared__ == wave-relative LDS byte offset.
HD unsigned lds_off(const void* p) { return (unsigned)(unsigned long long)p; }

// Async global->LDS 16B copy (per-lane addresses), tracked by ASYNCcnt.
HD void async_g2l_b128(unsigned lds_byte, unsigned gofs_byte, unsigned long long gbase) {
    asm volatile("global_load_async_to_lds_b128 %0, %1, %2"
                 :: "v"(lds_byte), "v"(gofs_byte), "s"(gbase) : "memory");
}
HD void wait_async0() { asm volatile("s_wait_asynccnt 0x0" ::: "memory"); }

// ---- WMMA fragment loaders (wave32), contiguous k = 16h + s in both A and B ----
// A (16x32 f16): lane L holds row M=L%16.  lanes 0-15: K = k0+0..15,
// lanes 16-31: K = k0+16..31.  Two 16B runs per lane.
// Works for both global and LDS-resident tiles (addrspace inferred).
HD v16h load_a_frag(const _Float16* __restrict__ A, int row, int ld, int k0, int lane) {
    int kb = (lane < 16) ? 0 : 16;
    const _Float16* p = A + (size_t)row * ld + k0 + kb;
    v8h lo = *(const v8h*)(p);
    v8h hi = *(const v8h*)(p + 8);
    v16h f;
#pragma unroll
    for (int i = 0; i < 8; ++i) { f[i] = lo[i]; f[8 + i] = hi[i]; }
    return f;
}

// B (32x16 f16): lane L holds column N=L%16; lanes 0-15: K=k0+0..15,
// lanes 16-31: K=k0+16..31.  Bt[col][k] contiguous in k -> one 32B run per lane.
HD v16h load_b_frag(const _Float16* __restrict__ Bt, int col, int ld, int k0, int lane) {
    int koff = (lane < 16) ? 0 : 16;
    return *(const v16h*)(Bt + (size_t)col * ld + k0 + koff);
}

HD v8f wmma_f16(v16h a, v16h b, v8f c) {
    return __builtin_amdgcn_wmma_f32_16x16x32_f16(false, a, false, b, (short)0, c, false, false);
}

HD void atomicMaxF(float* addr, float v) {
    if (v >= 0.0f) atomicMax((int*)addr, __float_as_int(v));
    else           atomicMin((unsigned int*)addr, __float_as_uint(v));
}

HD float warp_sum(float v) {
#pragma unroll
    for (int o = 16; o > 0; o >>= 1) v += __shfl_down(v, o);
    return v;   // lane 0 holds sum
}

// ================= elementwise / prep kernels =================
__global__ void k_fill(float* p, size_t n, float v) {
    size_t i = (size_t)blockIdx.x * blockDim.x + threadIdx.x;
    if (i < n) p[i] = v;
}

__global__ void k_cvt_f16(const float* __restrict__ s, _Float16* __restrict__ d, size_t n) {
    size_t i = (size_t)blockIdx.x * blockDim.x + threadIdx.x;
    if (i < n) d[i] = (_Float16)s[i];
}

// Wmsg (K,d,o) f32 -> Wt (K,o,d) f16  (so each output column is K-contiguous)
__global__ void k_transpose_wmsg(const float* __restrict__ W, _Float16* __restrict__ Wt) {
    size_t total = (size_t)N_ETYPE * D_H1 * D_H1;
    size_t i = (size_t)blockIdx.x * blockDim.x + threadIdx.x;
    if (i >= total) return;
    int k = (int)(i / (D_H1 * D_H1));
    int rem = (int)(i % (D_H1 * D_H1));
    int o = rem / D_H1, d = rem % D_H1;
    Wt[i] = (_Float16)W[(size_t)k * D_H1 * D_H1 + (size_t)d * D_H1 + o];
}

__global__ void k_init_h(const float* __restrict__ in_feat, float* __restrict__ h,
                         _Float16* __restrict__ h16) {
    size_t i = (size_t)blockIdx.x * blockDim.x + threadIdx.x;
    if (i >= (size_t)N_NODES * D_H1) return;
    int n = (int)(i / D_H1), c = (int)(i % D_H1);
    float v = (c < D_IN) ? in_feat[(size_t)n * D_IN + c] : 0.0f;
    h[i] = v; h16[i] = (_Float16)v;
}

// ================= f16 WMMA GEMM: C[16b, 512] = A-tile * Bt^T (+bias) =================
// Grid: one block per 16-row tile; 8 waves = 8 column strips of 64.
// A tile async-staged to LDS once (shared by all 8 waves); B double-buffered in regs.
__global__ void k_gemm_f16(const _Float16* __restrict__ A, const _Float16* __restrict__ Bt,
                           float* __restrict__ C, const float* __restrict__ bias,
                           int Kdim, int ldc) {
    __shared__ _Float16 As[16 * D_H1];
    int lane = threadIdx.x & 31;
    int strip = threadIdx.x >> 5;
    int rowBase = blockIdx.x * 16;
    int colBase = strip * 64;

    // cooperative async stage: tile is one contiguous (16*Kdim*2)-byte region
    {
        unsigned long long g = (unsigned long long)(A + (size_t)rowBase * Kdim);
        unsigned lb = lds_off(As);
        int chunks = (16 * Kdim * 2) / 16;
        for (int c = threadIdx.x; c < chunks; c += 256) {
            unsigned b = (unsigned)c * 16u;
            async_g2l_b128(lb + b, b, g);
        }
        wait_async0();
        __syncthreads();
    }

    v8f acc[4] = {};
    int cc = lane & 15;
    v16h b0[4], b1[4];
#pragma unroll
    for (int t = 0; t < 4; ++t) b0[t] = load_b_frag(Bt, colBase + t * 16 + cc, Kdim, 0, lane);
    for (int k0 = 0; k0 < Kdim; k0 += 64) {
#pragma unroll
        for (int t = 0; t < 4; ++t) b1[t] = load_b_frag(Bt, colBase + t * 16 + cc, Kdim, k0 + 32, lane);
        v16h a = load_a_frag(As, cc, Kdim, k0, lane);
#pragma unroll
        for (int t = 0; t < 4; ++t) acc[t] = wmma_f16(a, b0[t], acc[t]);
        if (k0 + 64 < Kdim) {
#pragma unroll
            for (int t = 0; t < 4; ++t) b0[t] = load_b_frag(Bt, colBase + t * 16 + cc, Kdim, k0 + 64, lane);
        }
        a = load_a_frag(As, cc, Kdim, k0 + 32, lane);
#pragma unroll
        for (int t = 0; t < 4; ++t) acc[t] = wmma_f16(a, b1[t], acc[t]);
    }

    // C/D layout: VGPR r -> M = r (lanes 0-15) or 8+r (lanes 16-31); N = lane%16.
    int rofs = (lane < 16) ? 0 : 8;
#pragma unroll
    for (int t = 0; t < 4; ++t) {
        int col = colBase + t * 16 + cc;
        float bv = bias ? bias[col] : 0.0f;
#pragma unroll
        for (int r = 0; r < 8; ++r)
            C[(size_t)(rowBase + rofs + r) * ldc + col] = acc[t][r] + bv;
    }
}

// ================= edge scatter for message passing =================
__global__ void k_scatter_msg(const float* __restrict__ lin, const int* __restrict__ src,
                              const int* __restrict__ dst, const int* __restrict__ et,
                              int ksel, float* __restrict__ acc) {
    int wave = (int)((blockIdx.x * blockDim.x + threadIdx.x) >> 5);
    int lane = threadIdx.x & 31;
    if (wave >= N_EDGES) return;
    if (et[wave] != ksel) return;
    const float* ps = lin + (size_t)src[wave] * D_H1;
    float* pd = acc + (size_t)dst[wave] * D_H1;
#pragma unroll
    for (int i = 0; i < 16; ++i) {
        int c = lane + 32 * i;
        atomicAdd(pd + c, ps[c]);
    }
}

// ================= fused GRU cell (6 WMMA streams, gates in registers) =================
// Grid: 4 blocks per 16-row tile; 8 waves/block = 8 of 32 column tiles.
// a16/h16 row tiles async-staged to LDS (shared by all 8 waves).
__global__ void k_gru(const _Float16* __restrict__ a16, const _Float16* __restrict__ h16,
                      const float* __restrict__ h_old,
                      const _Float16* __restrict__ wih, const _Float16* __restrict__ whh,
                      const float* __restrict__ b_ih, const float* __restrict__ b_hh,
                      float* __restrict__ h_new, _Float16* __restrict__ h16_new) {
    __shared__ _Float16 Sa[16 * D_H1];
    __shared__ _Float16 Sh[16 * D_H1];
    int lane = threadIdx.x & 31;
    int w = threadIdx.x >> 5;
    int rowBase = (blockIdx.x >> 2) * 16;
    int j = (((blockIdx.x & 3) << 3) + w) * 16;

    {
        unsigned long long ga = (unsigned long long)(a16 + (size_t)rowBase * D_H1);
        unsigned long long gh = (unsigned long long)(h16 + (size_t)rowBase * D_H1);
        unsigned la = lds_off(Sa), lh = lds_off(Sh);
        for (int c = threadIdx.x; c < (16 * D_H1 * 2) / 16; c += 256) {
            unsigned b = (unsigned)c * 16u;
            async_g2l_b128(la + b, b, ga);
            async_g2l_b128(lh + b, b, gh);
        }
        wait_async0();
        __syncthreads();
    }

    v8f ir = {}, iz = {}, in_ = {}, hr = {}, hz = {}, hn = {};
    int cc = lane & 15;
    for (int k0 = 0; k0 < D_H1; k0 += 32) {
        v16h af = load_a_frag(Sa, cc, D_H1, k0, lane);
        v16h hf = load_a_frag(Sh, cc, D_H1, k0, lane);
        int col = j + cc;
        v16h b[6];
        b[0] = load_b_frag(wih, col,            D_H1, k0, lane);
        b[1] = load_b_frag(wih, col + D_H1,     D_H1, k0, lane);
        b[2] = load_b_frag(wih, col + 2 * D_H1, D_H1, k0, lane);
        b[3] = load_b_frag(whh, col,            D_H1, k0, lane);
        b[4] = load_b_frag(whh, col + D_H1,     D_H1, k0, lane);
        b[5] = load_b_frag(whh, col + 2 * D_H1, D_H1, k0, lane);
        ir  = wmma_f16(af, b[0], ir);
        iz  = wmma_f16(af, b[1], iz);
        in_ = wmma_f16(af, b[2], in_);
        hr  = wmma_f16(hf, b[3], hr);
        hz  = wmma_f16(hf, b[4], hz);
        hn  = wmma_f16(hf, b[5], hn);
    }
    int rofs = (lane < 16) ? 0 : 8;
    int col = j + cc;
    float bir = b_ih[col], biz = b_ih[col + D_H1], bin = b_ih[col + 2 * D_H1];
    float bhr = b_hh[col], bhz = b_hh[col + D_H1], bhn = b_hh[col + 2 * D_H1];
#pragma unroll
    for (int r = 0; r < 8; ++r) {
        int m = rowBase + rofs + r;
        float rg = sigmoid_((ir[r] + bir) + (hr[r] + bhr));
        float zg = sigmoid_((iz[r] + biz) + (hz[r] + bhz));
        float ng = tanhf((in_[r] + bin) + rg * (hn[r] + bhn));
        float ho = h_old[(size_t)m * D_H1 + col];
        float hv = (1.0f - zg) * ng + zg * ho;
        h_new[(size_t)m * D_H1 + col] = hv;
        h16_new[(size_t)m * D_H1 + col] = (_Float16)hv;
    }
}

// ================= L2-normalize + sigmoid (warp per node) =================
__global__ void k_norm_sig(float* __restrict__ h, _Float16* __restrict__ h16) {
    int wave = (int)((blockIdx.x * blockDim.x + threadIdx.x) >> 5);
    int lane = threadIdx.x & 31;
    if (wave >= N_NODES) return;
    float* p = h + (size_t)wave * D_H1;
    float ss = 0.0f;
#pragma unroll
    for (int i = 0; i < 16; ++i) { float v = p[lane + 32 * i]; ss += v * v; }
    ss = warp_sum(ss);
    ss = __shfl(ss, 0);
    float inv = 1.0f / fmaxf(sqrtf(ss), 1e-12f);
#pragma unroll
    for (int i = 0; i < 16; ++i) {
        int c = lane + 32 * i;
        float v = sigmoid_(p[c] * inv);
        p[c] = v;
        h16[(size_t)wave * D_H1 + c] = (_Float16)v;
    }
}

// ================= GAT attention =================
__global__ void k_el_er(const float* __restrict__ feat, const float* __restrict__ al,
                        const float* __restrict__ ar, float* __restrict__ el,
                        float* __restrict__ er) {
    int wave = (int)((blockIdx.x * blockDim.x + threadIdx.x) >> 5);
    int lane = threadIdx.x & 31;
    if (wave >= N_NODES * N_HEADS) return;
    int n = wave >> 1, head = wave & 1;
    const float* pf = feat + (size_t)n * (N_HEADS * D_H2) + head * D_H2;
    float sl = 0.0f, sr = 0.0f;
#pragma unroll
    for (int i = 0; i < 8; ++i) {
        int c = lane + 32 * i;
        float f = pf[c];
        sl += f * al[head * D_H2 + c];
        sr += f * ar[head * D_H2 + c];
    }
    sl = warp_sum(sl); sr = warp_sum(sr);
    if (lane == 0) { el[wave] = sl; er[wave] = sr; }
}

__global__ void k_edge_score(const float* __restrict__ el, const float* __restrict__ er,
                             const int* __restrict__ src, const int* __restrict__ dst,
                             float* __restrict__ ebuf, float* __restrict__ emax) {
    int i = blockIdx.x * blockDim.x + threadIdx.x;
    if (i >= N_EDGES * N_HEADS) return;
    int e = i >> 1, h = i & 1;
    float s = el[src[e] * N_HEADS + h] + er[dst[e] * N_HEADS + h];
    s = (s > 0.0f) ? s : 0.2f * s;   // leaky_relu(0.2)
    ebuf[i] = s;
    atomicMaxF(&emax[dst[e] * N_HEADS + h], s);
}

__global__ void k_edge_exp(const float* __restrict__ emax, const int* __restrict__ dst,
                           float* __restrict__ ebuf, float* __restrict__ den) {
    int i = blockIdx.x * blockDim.x + threadIdx.x;
    if (i >= N_EDGES * N_HEADS) return;
    int e = i >> 1, h = i & 1;
    float ee = expf(ebuf[i] - emax[dst[e] * N_HEADS + h]);
    ebuf[i] = ee;
    atomicAdd(&den[dst[e] * N_HEADS + h], ee);
}

__global__ void k_edge_agg(const float* __restrict__ ebuf, const float* __restrict__ den,
                           const int* __restrict__ src, const int* __restrict__ dst,
                           const float* __restrict__ feat, float* __restrict__ rst) {
    int wave = (int)((blockIdx.x * blockDim.x + threadIdx.x) >> 5);
    int lane = threadIdx.x & 31;
    if (wave >= N_EDGES) return;
    int d = dst[wave], s = src[wave];
    float a0 = ebuf[wave * 2 + 0] / den[d * 2 + 0];
    float a1 = ebuf[wave * 2 + 1] / den[d * 2 + 1];
    const float* pf = feat + (size_t)s * D_H1;
    float* pr = rst + (size_t)d * D_H1;
#pragma unroll
    for (int i = 0; i < 16; ++i) {
        int c = lane + 32 * i;
        float alpha = (c < D_H2) ? a0 : a1;
        atomicAdd(pr + c, alpha * pf[c]);
    }
}

__global__ void k_graphsum(const float* __restrict__ rst, const float* __restrict__ gbias,
                           const int* __restrict__ gid, float* __restrict__ hg) {
    size_t i = (size_t)blockIdx.x * blockDim.x + threadIdx.x;
    if (i >= (size_t)N_NODES * D_H1) return;
    int n = (int)(i / D_H1), c = (int)(i % D_H1);
    float v = fmaxf(rst[i] + gbias[c], 0.0f);   // +bias, ReLU
    atomicAdd(&hg[(size_t)gid[n] * D_H1 + c], v);
}

__global__ void k_count(const int* __restrict__ gid, float* __restrict__ cnt) {
    int i = blockIdx.x * blockDim.x + threadIdx.x;
    if (i >= N_NODES) return;
    atomicAdd(&cnt[gid[i]], 1.0f);
}

__global__ void k_classify(const float* __restrict__ hg, const float* __restrict__ cnt,
                           const float* __restrict__ cw, const float* __restrict__ cb,
                           float* __restrict__ out) {
    int wave = (int)((blockIdx.x * blockDim.x + threadIdx.x) >> 5);
    int lane = threadIdx.x & 31;
    if (wave >= N_GRAPH * N_HEADS * N_CLS) return;
    int g = wave >> 5, head = (wave >> 4) & 1, c = wave & 15;
    const float* ph = hg + (size_t)g * D_H1 + head * D_H2;
    const float* pw = cw + (size_t)c * D_H2;
    float s = 0.0f;
#pragma unroll
    for (int i = 0; i < 8; ++i) { int d = lane + 32 * i; s += ph[d] * pw[d]; }
    s = warp_sum(s);
    if (lane == 0) out[wave] = s / fmaxf(cnt[g], 1.0f) + cb[c];
}

// =========================== host orchestration ===========================
static inline int blk1d(size_t n) { return (int)((n + 255) / 256); }
static inline int blkw(size_t waves) { return (int)((waves * 32 + 255) / 256); }

extern "C" void kernel_launch(void* const* d_in, const int* in_sizes, int n_in,
                              void* d_out, int out_size, void* d_ws, size_t ws_size,
                              hipStream_t stream) {
    (void)in_sizes; (void)n_in; (void)out_size; (void)ws_size;
    const float* in_feat  = (const float*)d_in[0];
    const int*   src      = (const int*)d_in[1];
    const int*   dst      = (const int*)d_in[2];
    const int*   etype    = (const int*)d_in[3];
    const int*   gid      = (const int*)d_in[4];
    const float* Wmsg     = (const float*)d_in[5];
    const float* bmsg     = (const float*)d_in[6];
    const float* w_ih     = (const float*)d_in[7];
    const float* w_hh     = (const float*)d_in[8];
    const float* b_ih     = (const float*)d_in[9];
    const float* b_hh     = (const float*)d_in[10];
    const float* fc_w     = (const float*)d_in[11];
    const float* attn_l   = (const float*)d_in[12];
    const float* attn_r   = (const float*)d_in[13];
    const float* gat_bias = (const float*)d_in[14];
    const float* cw       = (const float*)d_in[15];
    const float* cb       = (const float*)d_in[16];
    float* out = (float*)d_out;

    const size_t NH = (size_t)N_NODES * D_H1;
    char* base = (char*)d_ws; size_t off = 0;
    auto carve = [&](size_t bytes) -> void* {
        void* p = base + off; off += (bytes + 255) & ~(size_t)255; return p;
    };
    float*    hA    = (float*)carve(NH * 4);
    float*    hB    = (float*)carve(NH * 4);
    _Float16* h16A  = (_Float16*)carve(NH * 2);
    _Float16* h16B  = (_Float16*)carve(NH * 2);
    float*    aF    = (float*)carve(NH * 4);          // msg accum; reused as rst
    _Float16* a16   = (_Float16*)carve(NH * 2);
    float*    tmp   = (float*)carve(NH * 4);          // per-etype lin; reused as feat
    _Float16* wmsgT = (_Float16*)carve((size_t)N_ETYPE * D_H1 * D_H1 * 2);
    _Float16* wih16 = (_Float16*)carve((size_t)3 * D_H1 * D_H1 * 2);
    _Float16* whh16 = (_Float16*)carve((size_t)3 * D_H1 * D_H1 * 2);
    _Float16* fc16  = (_Float16*)carve((size_t)D_H1 * D_H1 * 2);
    float*    ebuf  = (float*)carve((size_t)N_EDGES * N_HEADS * 4);
    float*    el    = (float*)carve((size_t)N_NODES * N_HEADS * 4);
    float*    er    = (float*)carve((size_t)N_NODES * N_HEADS * 4);
    float*    emax  = (float*)carve((size_t)N_NODES * N_HEADS * 4);
    float*    den   = (float*)carve((size_t)N_NODES * N_HEADS * 4);
    float*    hg    = (float*)carve((size_t)N_GRAPH * D_H1 * 4);
    float*    cnt   = (float*)carve((size_t)N_GRAPH * 4);

    const int Mtiles = N_NODES / 16;   // 1250

    // ---- weight prep (f16, column-K-contiguous) ----
    k_transpose_wmsg<<<blk1d((size_t)N_ETYPE * D_H1 * D_H1), 256, 0, stream>>>(Wmsg, wmsgT);
    k_cvt_f16<<<blk1d((size_t)3 * D_H1 * D_H1), 256, 0, stream>>>(w_ih, wih16, (size_t)3 * D_H1 * D_H1);
    k_cvt_f16<<<blk1d((size_t)3 * D_H1 * D_H1), 256, 0, stream>>>(w_hh, whh16, (size_t)3 * D_H1 * D_H1);
    k_cvt_f16<<<blk1d((size_t)D_H1 * D_H1), 256, 0, stream>>>(fc_w, fc16, (size_t)D_H1 * D_H1);

    // ---- init h (pad 256 -> 512) ----
    k_init_h<<<blk1d(NH), 256, 0, stream>>>(in_feat, hA, h16A);
    float* h_cur = hA;  float* h_nxt = hB;
    _Float16* h16_cur = h16A;  _Float16* h16_nxt = h16B;

    // ---- GatedGraphConv steps ----
    for (int step = 0; step < N_STEPS; ++step) {
        k_fill<<<blk1d(NH), 256, 0, stream>>>(aF, NH, 0.0f);
        for (int k = 0; k < N_ETYPE; ++k) {
            k_gemm_f16<<<Mtiles, 256, 0, stream>>>(
                h16_cur, wmsgT + (size_t)k * D_H1 * D_H1, tmp, bmsg + (size_t)k * D_H1,
                D_H1, D_H1);
            k_scatter_msg<<<blkw(N_EDGES), 256, 0, stream>>>(tmp, src, dst, etype, k, aF);
        }
        k_cvt_f16<<<blk1d(NH), 256, 0, stream>>>(aF, a16, NH);
        k_gru<<<Mtiles * 4, 256, 0, stream>>>(
            a16, h16_cur, h_cur, wih16, whh16, b_ih, b_hh, h_nxt, h16_nxt);
        { float* t = h_cur; h_cur = h_nxt; h_nxt = t; }
        { _Float16* t = h16_cur; h16_cur = h16_nxt; h16_nxt = t; }
    }

    // ---- normalize + sigmoid ----
    k_norm_sig<<<blkw(N_NODES), 256, 0, stream>>>(h_cur, h16_cur);

    // ---- GAT: feat = h @ fc_w^T ----
    k_gemm_f16<<<Mtiles, 256, 0, stream>>>(
        h16_cur, fc16, tmp, (const float*)nullptr, D_H1, D_H1);
    k_el_er<<<blkw((size_t)N_NODES * N_HEADS), 256, 0, stream>>>(tmp, attn_l, attn_r, el, er);

    k_fill<<<blk1d((size_t)N_NODES * N_HEADS), 256, 0, stream>>>(emax, (size_t)N_NODES * N_HEADS, -1e30f);
    k_fill<<<blk1d((size_t)N_NODES * N_HEADS), 256, 0, stream>>>(den, (size_t)N_NODES * N_HEADS, 0.0f);
    k_edge_score<<<blk1d((size_t)N_EDGES * N_HEADS), 256, 0, stream>>>(el, er, src, dst, ebuf, emax);
    k_edge_exp<<<blk1d((size_t)N_EDGES * N_HEADS), 256, 0, stream>>>(emax, dst, ebuf, den);

    float* rst = aF;   // reuse
    k_fill<<<blk1d(NH), 256, 0, stream>>>(rst, NH, 0.0f);
    k_edge_agg<<<blkw(N_EDGES), 256, 0, stream>>>(ebuf, den, src, dst, tmp, rst);

    // ---- graph mean + classify ----
    k_fill<<<blk1d((size_t)N_GRAPH * D_H1), 256, 0, stream>>>(hg, (size_t)N_GRAPH * D_H1, 0.0f);
    k_fill<<<blk1d(N_GRAPH), 256, 0, stream>>>(cnt, N_GRAPH, 0.0f);
    k_graphsum<<<blk1d(NH), 256, 0, stream>>>(rst, gat_bias, gid, hg);
    k_count<<<blk1d(N_NODES), 256, 0, stream>>>(gid, cnt);
    k_classify<<<blkw((size_t)N_GRAPH * N_HEADS * N_CLS), 256, 0, stream>>>(hg, cnt, cw, cb, out);
}
